// ExodusNet_86079734546562
// MI455X (gfx1250) — matrix operand order repaired and checked by reading, not compiled, not run on hardware
//
#include <hip/hip_runtime.h>

typedef __attribute__((ext_vector_type(16))) __bf16 v16bf;
typedef __attribute__((ext_vector_type(8)))  float  v8f;

#define ALPHA 0.951229424500714f   // exp(-1/20)
#define NORM  0.048770575499286f   // 1 - ALPHA

#define KDIM  700
#define KPAD  704                  // 11 * 64
#define TK    64                   // K-slice per double-buffer stage
#define LDT   72                   // padded LDS row stride (bf16 elems) = 36 dwords

__device__ __forceinline__ unsigned short f2bf(float f) {
  unsigned u = __builtin_bit_cast(unsigned int, f);
  u += 0x7FFFu + ((u >> 16) & 1u);   // round-to-nearest-even
  return (unsigned short)(u >> 16);
}

union FragBF { uint4 q[2]; v16bf v; };

// ---------------------------------------------------------------------------
// Pass 0: fp32 -> bf16 with K padded 700 -> 704 (zero fill). 4 elems/thread.
// ---------------------------------------------------------------------------
__global__ __launch_bounds__(256) void convert_pad_bf16(
    const float* __restrict__ src, unsigned short* __restrict__ dst,
    long long nquads)
{
  long long q = (long long)blockIdx.x * blockDim.x + threadIdx.x;
  if (q >= nquads) return;
  const int qpr = KPAD >> 2;                 // quads per row = 176
  long long r = q / qpr;
  int k = (int)(q - r * qpr) << 2;
  const float* s = src + r * KDIM + k;
  __align__(8) unsigned short t[4];
  if (k + 3 < KDIM) {
    float4 f = *(const float4*)s;            // 700 % 4 == 0 -> aligned
    t[0] = f2bf(f.x); t[1] = f2bf(f.y); t[2] = f2bf(f.z); t[3] = f2bf(f.w);
  } else {
    #pragma unroll
    for (int i = 0; i < 4; ++i)
      t[i] = (k + i < KDIM) ? f2bf(s[i]) : (unsigned short)0;
  }
  *(uint2*)&dst[r * KPAD + k] = *(const uint2*)t;
}

// ---------------------------------------------------------------------------
// Pass 1: C[M,N] = A[M,K] * B[N,K]^T, bf16 in (K padded), fp32 out.
// 128x128 block tile, 8 waves of 64x32, TK=64, double-buffered async->LDS.
// ---------------------------------------------------------------------------
__global__ __launch_bounds__(256) void gemm1_wmma(
    const unsigned short* __restrict__ A, const unsigned short* __restrict__ B,
    float* __restrict__ C, int M, int N)
{
  __shared__ __align__(16) unsigned short As[2][128 * LDT];
  __shared__ __align__(16) unsigned short Bs[2][128 * LDT];

  const int tid  = threadIdx.x;
  const int lane = tid & 31;
  const int wave = tid >> 5;

  const int m0 = blockIdx.y * 128;
  const int n0 = blockIdx.x * 128;

  // 8 waves: 2 wave-rows (64 M) x 4 wave-cols (32 N)
  const int mW = (wave & 1) * 64;
  const int nW = (wave >> 1) * 32;

  const int laneRow = lane & 15;
  const int aK = (lane >> 4) * 8;    // A frag: K chunks at aK, aK+16 (per 32-K subtile)
  const int bK = (lane >> 4) * 8;    // B frag: K chunks at bK, bK+16 (per 32-K subtile)

  v8f acc[4][2] = {};

  // per-thread async chunk coords: 4 b128 chunks per tile (1024 chunks / 256 thr)
  const int cRow = tid >> 3;          // base row 0..31 (j adds 32 each)
  const int cOff = (tid & 7) * 16;    // byte offset within 128B row slice

  const unsigned ldsA0 = (unsigned)(size_t)&As[0][0];
  const unsigned ldsA1 = (unsigned)(size_t)&As[1][0];
  const unsigned ldsB0 = (unsigned)(size_t)&Bs[0][0];
  const unsigned ldsB1 = (unsigned)(size_t)&Bs[1][0];

  const char* gA = (const char*)(A + (size_t)m0 * KPAD);
  const char* gB = (const char*)(B + (size_t)n0 * KPAD);

  #define ISSUE_TILE(LA, LB, K0)                                              \
    {                                                                         \
      _Pragma("unroll")                                                       \
      for (int j = 0; j < 4; ++j) {                                           \
        int row = cRow + j * 32;                                              \
        unsigned la = (LA) + (unsigned)(row * (LDT * 2) + cOff);              \
        unsigned lb = (LB) + (unsigned)(row * (LDT * 2) + cOff);              \
        const char* pa = gA + (size_t)row * (KPAD * 2) + (K0) * 2 + cOff;     \
        const char* pb = gB + (size_t)row * (KPAD * 2) + (K0) * 2 + cOff;     \
        asm volatile("global_load_async_to_lds_b128 %0, %1, off"              \
                     :: "v"(la), "v"(pa) : "memory");                         \
        asm volatile("global_load_async_to_lds_b128 %0, %1, off"              \
                     :: "v"(lb), "v"(pb) : "memory");                         \
      }                                                                       \
    }

  ISSUE_TILE(ldsA0, ldsB0, 0)                 // prologue: stage tile 0

  const int NITER = KPAD / TK;                // 11
  for (int ki = 0; ki < NITER; ++ki) {
    const int buf = ki & 1;
    if (ki + 1 < NITER) {
      // stage next tile into the other buffer, then wait for current tile
      if (buf == 0) { ISSUE_TILE(ldsA1, ldsB1, (ki + 1) * TK) }
      else          { ISSUE_TILE(ldsA0, ldsB0, (ki + 1) * TK) }
      asm volatile("s_wait_asynccnt 0x8" ::: "memory");
    } else {
      asm volatile("s_wait_asynccnt 0x0" ::: "memory");
    }
    __syncthreads();

    const unsigned short* At = As[buf];
    const unsigned short* Bt = Bs[buf];
    #pragma unroll
    for (int ks = 0; ks < 2; ++ks) {          // two 32-wide K subtiles
      const int kb = ks * 32;
      FragBF a[4], b[2];
      #pragma unroll
      for (int mf = 0; mf < 4; ++mf) {
        int row = mW + mf * 16 + laneRow;
        a[mf].q[0] = *(const uint4*)&At[row * LDT + kb + aK];
        a[mf].q[1] = *(const uint4*)&At[row * LDT + kb + aK + 16];
      }
      #pragma unroll
      for (int nf = 0; nf < 2; ++nf) {
        int col = nW + nf * 16 + laneRow;
        b[nf].q[0] = *(const uint4*)&Bt[col * LDT + kb + bK];
        b[nf].q[1] = *(const uint4*)&Bt[col * LDT + kb + bK + 16];
      }
      #pragma unroll
      for (int mf = 0; mf < 4; ++mf)
        #pragma unroll
        for (int nf = 0; nf < 2; ++nf)
          acc[mf][nf] = __builtin_amdgcn_wmma_f32_16x16x32_bf16(
              false, a[mf].v, false, b[nf].v,
              (short)0, acc[mf][nf], false, false);
    }
    __syncthreads();
  }
  #undef ISSUE_TILE

  // C/D layout: VGPR r -> M = r (+8 for lanes 16-31), lane&15 -> N
  #pragma unroll
  for (int mf = 0; mf < 4; ++mf) {
    #pragma unroll
    for (int nf = 0; nf < 2; ++nf) {
      int n     = n0 + nW + nf * 16 + laneRow;
      int mBase = m0 + mW + mf * 16 + (lane >> 4) * 8;
      #pragma unroll
      for (int r = 0; r < 8; ++r)
        C[(size_t)(mBase + r) * N + n] = acc[mf][nf][r];
    }
  }
}

// ---------------------------------------------------------------------------
// Pass 2: fused LIF1 over T -> dot over H with W2 -> LIF2 -> spikes (B,T,1)
// ---------------------------------------------------------------------------
__global__ __launch_bounds__(512) void lif_fused(
    const float* __restrict__ w1, const float* __restrict__ W2,
    float* __restrict__ out, int T, int H)
{
  const int b    = blockIdx.x;
  const int tid  = threadIdx.x;    // == h channel
  const int lane = tid & 31;
  const int wave = tid >> 5;       // 0..15

  __shared__ float red[4][16];

  const float w2h = W2[tid];
  float v1 = 0.0f;
  float v2 = 0.0f;

  const float* base = w1 + (size_t)b * T * H + tid;

  for (int t0 = 0; t0 < T; t0 += 4) {
    float c[4];
    #pragma unroll
    for (int j = 0; j < 4; ++j) {
      float w = base[(size_t)(t0 + j) * H];      // coalesced across h
      v1 = ALPHA * v1 + NORM * w;
      float s = (v1 >= 1.0f) ? 1.0f : 0.0f;      // SingleSpike
      v1 -= s;                                    // membrane-subtract reset
      c[j] = s * w2h;
    }
    #pragma unroll
    for (int j = 0; j < 4; ++j) {
      #pragma unroll
      for (int off = 16; off >= 1; off >>= 1)
        c[j] += __shfl_xor(c[j], off, 32);
    }
    if (lane == 0) {
      #pragma unroll
      for (int j = 0; j < 4; ++j) red[j][wave] = c[j];
    }
    __syncthreads();
    if (tid == 0) {
      #pragma unroll
      for (int j = 0; j < 4; ++j) {
        float sum = 0.0f;
        #pragma unroll
        for (int wv = 0; wv < 16; ++wv) sum += red[j][wv];
        v2 = ALPHA * v2 + NORM * sum;
        float s2 = (v2 >= 1.0f) ? 1.0f : 0.0f;
        v2 -= s2;
        out[(size_t)b * T + t0 + j] = s2;
      }
    }
    __syncthreads();
  }
}

extern "C" void kernel_launch(void* const* d_in, const int* in_sizes, int n_in,
                              void* d_out, int out_size, void* d_ws, size_t ws_size,
                              hipStream_t stream) {
  const float* x  = (const float*)d_in[0];   // (256, 500, 700)
  const float* W1 = (const float*)d_in[1];   // (512, 700)
  const float* W2 = (const float*)d_in[2];   // (1, 512)
  float* out = (float*)d_out;                // (256, 500, 1)

  const int B = 256, T = 500, H = 512;
  const int M = B * T;                       // 128000

  // workspace layout (all 16B aligned)
  char* ws = (char*)d_ws;
  unsigned short* xbf  = (unsigned short*)ws;                        // M*KPAD bf16 = 180,224,000 B
  unsigned short* w1bf = (unsigned short*)(ws + (size_t)M * KPAD * 2);           // H*KPAD bf16
  float*          w1f  = (float*)(ws + (size_t)M * KPAD * 2 + (size_t)H * KPAD * 2); // M*H fp32

  // pass 0: convert + pad
  long long qx = (long long)M * (KPAD / 4);
  convert_pad_bf16<<<(unsigned)((qx + 255) / 256), 256, 0, stream>>>(x, xbf, qx);
  long long qw = (long long)H * (KPAD / 4);
  convert_pad_bf16<<<(unsigned)((qw + 255) / 256), 256, 0, stream>>>(W1, w1bf, qw);

  // pass 1: WMMA GEMM
  dim3 grid1(H / 128, M / 128);              // (4, 1000)
  gemm1_wmma<<<grid1, 256, 0, stream>>>(xbf, w1bf, w1f, M, H);

  // pass 2: fused LIF -> dot -> LIF
  lif_fused<<<B, H, 0, stream>>>(w1f, W2, out, T, H);
}
